// leg_actor_28690381537989
// MI455X (gfx1250) — matrix operand order repaired and checked by reading, not compiled, not run on hardware
//
#include <hip/hip_runtime.h>
#include <math.h>

typedef __attribute__((ext_vector_type(16))) _Float16 v16h;
typedef __attribute__((ext_vector_type(8)))  _Float16 v8h;
typedef __attribute__((ext_vector_type(8)))  float    v8f;

#define WG_THREADS 128      // 4 waves (wave32)
#define NG 12               // groups per workgroup
#define RROWS (NG*5)        // 60 real rows
#define RPAD 64             // padded rows (4 waves x 16)
#define HS 72               // LDS row stride in halves (64 data + 8 pad, keeps 16B align)
#define SWSTRIDE 24         // per-lane halves in swizzled weight LDS (16 data + 8 pad)
#define SOFTPLUS_BIAS 0.5413248546129181f
#define SCALE_LB 1e-4f

// ---------------------------------------------------------------------------
// Prep: convert the four 64x64 f32 weight matrices into the f16 B-operand
// swizzle for v_wmma_f32_16x16x32_f16.
// B[k][n] = W[n][k]  (out = h @ W^T). Per (matrix m, ktile, ntile, lane):
// 16 contiguous halves: k = kt*32 + (lane>=16 ? 16:0) + s, n = nt*16 + (lane&15)
// Compact layout in d_ws: ((m*8 + kt*4 + nt)*32 + lane)*16 + s
// ---------------------------------------------------------------------------
__global__ void prep_weights(const float* __restrict__ Wroot1,
                             const float* __restrict__ Wrel1,
                             const float* __restrict__ Wroot2,
                             const float* __restrict__ Wrel2,
                             _Float16* __restrict__ wsw) {
    int t = blockIdx.x * blockDim.x + threadIdx.x;
    if (t >= 4 * 8 * 32 * 16) return;
    int s    = t & 15;
    int lane = (t >> 4) & 31;
    int tile = (t >> 9) & 7;        // kt*4 + nt
    int m    = (t >> 12) & 3;
    int kt = tile >> 2, nt = tile & 3;
    int k = kt * 32 + ((lane >= 16) ? 16 : 0) + s;
    int n = nt * 16 + (lane & 15);
    const float* W = (m == 0) ? Wroot1 : (m == 1) ? Wrel1 : (m == 2) ? Wroot2 : Wrel2;
    wsw[t] = (_Float16)W[n * 64 + k];
}

// ---------------------------------------------------------------------------
// Main kernel: 12 groups (60 nodes) per workgroup, everything in LDS.
// ---------------------------------------------------------------------------
__global__ __launch_bounds__(WG_THREADS)
void leg_actor_main(const float* __restrict__ x,
                    const float* __restrict__ Wt, const float* __restrict__ bt,
                    const float* __restrict__ Wj, const float* __restrict__ bj,
                    const float* __restrict__ brel1,
                    const float* __restrict__ brel2,
                    const _Float16* __restrict__ wsw,
                    const float* __restrict__ Wo, const float* __restrict__ bo,
                    float* __restrict__ out, int B) {
    // 2 matrices (root, rel) per layer, 8 tiles each, 32 lanes, SWSTRIDE halves
    __shared__ __align__(16) _Float16 sW[2 * 8 * 32 * SWSTRIDE];   // 24 KB
    __shared__ __align__(16) _Float16 sH[RPAD * HS];               // 9 KB
    __shared__ __align__(16) _Float16 sQ[RPAD * HS];               // 9 KB
    __shared__ __align__(16) float sX[RROWS * 11];                 // 2.6 KB
    __shared__ float sWt[64 * 11], sbt[64], sWj[64 * 4], sbj[64];
    __shared__ float sbrel[2 * 64];
    __shared__ float sWo[16 * 64], sbo[16];

    const int tid   = threadIdx.x;
    const int gbase = blockIdx.x * NG;
    int gcnt = B - gbase; if (gcnt > NG) gcnt = NG;
    const int rcnt = gcnt * 5;

    // ---- stage weights for layer 0 + small params + x tile ----------------
    {   // layer-0 swizzled weights (Wroot1=m0, Wrel1=m1): 1024 x 16B granules
        for (int i = tid; i < 1024; i += WG_THREADS) {
            int mloc = i >> 9;              // 0..1
            int rest = i & 511;
            int tile = rest >> 6;           // 0..7
            int lane = (rest >> 1) & 31;
            int gr   = rest & 1;
            const v8h* src = (const v8h*)&wsw[((mloc * 8 + tile) * 32 + lane) * 16 + gr * 8];
            *(v8h*)&sW[((mloc * 8 + tile) * 32 + lane) * SWSTRIDE + gr * 8] = *src;
        }
    }
    for (int i = tid; i < 64 * 11; i += WG_THREADS) sWt[i] = Wt[i];
    for (int i = tid; i < 256; i += WG_THREADS) sWj[i] = Wj[i];
    for (int i = tid; i < 1024; i += WG_THREADS) sWo[i] = Wo[i];
    for (int i = tid; i < 64; i += WG_THREADS) {
        sbt[i] = bt[i]; sbj[i] = bj[i];
        sbrel[i] = brel1[i]; sbrel[64 + i] = brel2[i];
    }
    for (int i = tid; i < 16; i += WG_THREADS) sbo[i] = bo[i];
    {
        const float* xsrc = x + (size_t)gbase * 55;   // 5 nodes * 11 feats
        int nflt = rcnt * 11;
        for (int i = tid; i < nflt; i += WG_THREADS) sX[i] = xsrc[i];
    }
    __syncthreads();

    // ---- embedding: h0 into sH (f16), zero the pad rows -------------------
    for (int idx = tid; idx < RPAD * 64; idx += WG_THREADS) {
        int r = idx >> 6, c = idx & 63;
        float acc = 0.f;
        if (r < rcnt) {
            int pos = r % 5;
            const float* xr = &sX[r * 11];
            if (pos == 0) {
                acc = sbt[c];
                const float* w = &sWt[c * 11];
                #pragma unroll
                for (int f = 0; f < 11; ++f) acc += w[f] * xr[f];
            } else {
                acc = sbj[c];
                const float* w = &sWj[c * 4];
                #pragma unroll
                for (int f = 0; f < 4; ++f) acc += w[f] * xr[f];
            }
        }
        sH[r * HS + c] = (_Float16)acc;
    }
    __syncthreads();

    const int wv    = tid >> 5;
    const int lane  = tid & 31;
    const int rbase = wv * 16;
    const int arow  = rbase + (lane & 15);
    const int aoff  = (lane >= 16) ? 16 : 0;  // half offset for hi half-wave
    const int hiM   = (lane >= 16) ? 8 : 0;
    const int ncol  = lane & 15;

    for (int layer = 0; layer < 2; ++layer) {
        // ---- A operands: 16x32 f16 tiles, contiguous k = 16*(lane/16) + s ------------
        v16h A[2];
        #pragma unroll
        for (int kt = 0; kt < 2; ++kt) {
            int h0 = arow * HS + kt * 32 + aoff;
            v8h lo = *(const v8h*)&sH[h0];
            v8h hi = *(const v8h*)&sH[h0 + 8];
            v16h a;
            #pragma unroll
            for (int e = 0; e < 8; ++e) { a[e] = lo[e]; a[8 + e] = hi[e]; }
            A[kt] = a;
        }

        // ---- GEMMs: P = H*Wroot^T (regs), Q = H*Wrel^T (-> LDS f16) -------
        v8f P[4];
        #pragma unroll
        for (int nt = 0; nt < 4; ++nt) {
            v8f cP = {};
            v8f cQ = {};
            #pragma unroll
            for (int kt = 0; kt < 2; ++kt) {
                const _Float16* bp = &sW[((0 * 8 + kt * 4 + nt) * 32 + lane) * SWSTRIDE];
                const _Float16* bq = &sW[((1 * 8 + kt * 4 + nt) * 32 + lane) * SWSTRIDE];
                v8h plo = *(const v8h*)bp;   v8h phi = *(const v8h*)(bp + 8);
                v8h qlo = *(const v8h*)bq;   v8h qhi = *(const v8h*)(bq + 8);
                v16h bP, bQ;
                #pragma unroll
                for (int e = 0; e < 8; ++e) {
                    bP[e] = plo[e]; bP[8 + e] = phi[e];
                    bQ[e] = qlo[e]; bQ[8 + e] = qhi[e];
                }
                cP = __builtin_amdgcn_wmma_f32_16x16x32_f16(false, A[kt], false, bP,
                                                            (short)0, cP, false, false);
                cQ = __builtin_amdgcn_wmma_f32_16x16x32_f16(false, A[kt], false, bQ,
                                                            (short)0, cQ, false, false);
            }
            P[nt] = cP;
            // spill Q to LDS as f16 (C layout: VGPR v -> M=v(+8 hi lanes), N=lane&15)
            #pragma unroll
            for (int v = 0; v < 8; ++v) {
                int row = rbase + v + hiM;
                sQ[row * HS + nt * 16 + ncol] = (_Float16)cQ[v];
            }
        }
        __syncthreads();   // all Q written, all H reads done

        // ---- star aggregation + bias + tanh -> new H ----------------------
        const float* br = &sbrel[layer * 64];
        #pragma unroll
        for (int nt = 0; nt < 4; ++nt) {
            int col = nt * 16 + ncol;
            #pragma unroll
            for (int v = 0; v < 8; ++v) {
                int row = rbase + v + hiM;
                if (row < rcnt) {
                    int pos = row % 5;
                    int g5  = row - pos;
                    float q;
                    if (pos == 0) {
                        q = (float)sQ[(g5 + 1) * HS + col] + (float)sQ[(g5 + 2) * HS + col]
                          + (float)sQ[(g5 + 3) * HS + col] + (float)sQ[(g5 + 4) * HS + col];
                    } else {
                        q = (float)sQ[g5 * HS + col];
                    }
                    float val = tanhf(P[nt][v] + q + br[col]);
                    sH[row * HS + col] = (_Float16)val;
                }
            }
        }
        // stage layer-1 weights while we're between barriers
        if (layer == 0) {
            for (int i = tid; i < 1024; i += WG_THREADS) {
                int mloc = i >> 9;
                int rest = i & 511;
                int tile = rest >> 6;
                int ln   = (rest >> 1) & 31;
                int gr   = rest & 1;
                const v8h* src = (const v8h*)&wsw[(((2 + mloc) * 8 + tile) * 32 + ln) * 16 + gr * 8];
                *(v8h*)&sW[((mloc * 8 + tile) * 32 + ln) * SWSTRIDE + gr * 8] = *src;
            }
        }
        __syncthreads();
    }

    // ---- output head: per joint, 4 outputs; loc + clamped softplus scale --
    for (int idx = tid; idx < gcnt * 16; idx += WG_THREADS) {
        int bl  = idx >> 4;
        int rem = idx & 15;
        int i = rem >> 2, o = rem & 3;
        int row = bl * 5 + 1 + i;
        const float* w = &sWo[(i * 4 + o) * 64];
        const _Float16* hh = &sH[row * HS];
        float acc = sbo[i * 4 + o];
        #pragma unroll
        for (int f = 0; f < 64; ++f) acc += w[f] * (float)hh[f];
        size_t b = (size_t)gbase + bl;
        if (o < 2) {
            out[b * 8 + i * 2 + o] = acc;                       // loc
        } else {
            float z  = acc + SOFTPLUS_BIAS;
            float sp = (z > 20.f) ? z : log1pf(expf(z));
            out[(size_t)B * 8 + b * 8 + i * 2 + (o - 2)] = fmaxf(sp, SCALE_LB);
        }
    }
}

// ---------------------------------------------------------------------------
extern "C" void kernel_launch(void* const* d_in, const int* in_sizes, int n_in,
                              void* d_out, int out_size, void* d_ws, size_t ws_size,
                              hipStream_t stream) {
    const float* x      = (const float*)d_in[0];
    // d_in[1] = edge_index (fixed star topology; unused)
    const float* Wt     = (const float*)d_in[2];
    const float* bt     = (const float*)d_in[3];
    const float* Wj     = (const float*)d_in[4];
    const float* bj     = (const float*)d_in[5];
    const float* Wrel1  = (const float*)d_in[6];
    const float* brel1  = (const float*)d_in[7];
    const float* Wroot1 = (const float*)d_in[8];
    const float* Wrel2  = (const float*)d_in[9];
    const float* brel2  = (const float*)d_in[10];
    const float* Wroot2 = (const float*)d_in[11];
    const float* Wo     = (const float*)d_in[12];
    const float* bo     = (const float*)d_in[13];

    int B = in_sizes[0] / 55;                 // x is [5B, 11]
    _Float16* wsw = (_Float16*)d_ws;          // 32 KB swizzled f16 weights

    prep_weights<<<64, 256, 0, stream>>>(Wroot1, Wrel1, Wroot2, Wrel2, wsw);

    int nblk = (B + NG - 1) / NG;
    leg_actor_main<<<nblk, WG_THREADS, 0, stream>>>(
        x, Wt, bt, Wj, bj, brel1, brel2, wsw, Wo, bo, (float*)d_out, B);
}
